// GIN_17377437680137
// MI455X (gfx1250) — matrix operand order repaired and checked by reading, not compiled, not run on hardware
//
#include <hip/hip_runtime.h>

#define D        128
#define NNODES   50000
#define NEDGES   800000
#define NGRAPHS  256
#define NLAYERS  3

typedef __attribute__((ext_vector_type(16))) __bf16 v16bf;
typedef __attribute__((ext_vector_type(8)))  __bf16 v8bf;
typedef __attribute__((ext_vector_type(8)))  float  v8f;
typedef __attribute__((ext_vector_type(4)))  float  v4f;

// Native non-returning fp32 atomic add (device scope) — guaranteed
// global_atomic_add_f32, no CAS-loop fallback.
__device__ __forceinline__ void atomic_add_f32_dev(float* p, float v) {
  asm volatile("global_atomic_add_f32 %0, %1, off scope:SCOPE_DEV"
               :: "v"((unsigned long long)(uintptr_t)p), "v"(v)
               : "memory");
}

// ---------------------------------------------------------------------------
// Zero a buffer (16B granularity)
// ---------------------------------------------------------------------------
__global__ __launch_bounds__(256) void gin_zero4_kernel(v4f* __restrict__ p, int n4) {
  int i = blockIdx.x * 256 + threadIdx.x;
  int stride = gridDim.x * 256;
  v4f z = {0.f, 0.f, 0.f, 0.f};
  for (; i < n4; i += stride) p[i] = z;
}

// ---------------------------------------------------------------------------
// Pack W1/W2 (row-major [layer][k][n] f32) -> bf16 transposed [layer][n][k]
// ---------------------------------------------------------------------------
__global__ __launch_bounds__(256) void gin_pack_w_kernel(
    const float* __restrict__ W1, const float* __restrict__ W2,
    __bf16* __restrict__ wt1, __bf16* __restrict__ wt2) {
  int idx = blockIdx.x * 256 + threadIdx.x;
  if (idx >= NLAYERS * D * D) return;
  int l = idx / (D * D);
  int r = idx - l * D * D;
  int n = r / D;
  int k = r - n * D;
  wt1[(l * D + n) * D + k] = (__bf16)W1[(l * D + k) * D + n];
  wt2[(l * D + n) * D + k] = (__bf16)W2[(l * D + k) * D + n];
}

// ---------------------------------------------------------------------------
// CSR build (once per call; graph is constant across layers)
// ---------------------------------------------------------------------------
__global__ __launch_bounds__(256) void gin_count_kernel(
    const int* __restrict__ dst, int* __restrict__ deg) {
  int e = blockIdx.x * 256 + threadIdx.x;
  if (e < NEDGES) atomicAdd(&deg[dst[e]], 1);
}

// Per-block inclusive scan (Hillis-Steele in LDS) + block totals.
__global__ __launch_bounds__(256) void gin_scan_block_kernel(
    const int* __restrict__ deg, int* __restrict__ incl, int* __restrict__ blockSums) {
  __shared__ int s[256];
  int tid = threadIdx.x;
  int gid = blockIdx.x * 256 + tid;
  s[tid] = (gid < NNODES) ? deg[gid] : 0;
  __syncthreads();
  for (int off = 1; off < 256; off <<= 1) {
    int t = (tid >= off) ? s[tid - off] : 0;
    __syncthreads();
    s[tid] += t;
    __syncthreads();
  }
  if (gid < NNODES) incl[gid] = s[tid];
  if (tid == 255) blockSums[blockIdx.x] = s[255];
}

// Single-block inclusive scan of block sums (nb <= 256).
__global__ __launch_bounds__(256) void gin_scan_sums_kernel(int* __restrict__ bs, int nb) {
  __shared__ int s[256];
  int tid = threadIdx.x;
  s[tid] = (tid < nb) ? bs[tid] : 0;
  __syncthreads();
  for (int off = 1; off < 256; off <<= 1) {
    int t = (tid >= off) ? s[tid - off] : 0;
    __syncthreads();
    s[tid] += t;
    __syncthreads();
  }
  if (tid < nb) bs[tid] = s[tid];
}

__global__ __launch_bounds__(256) void gin_finalize_rowptr_kernel(
    const int* __restrict__ deg, const int* __restrict__ incl,
    const int* __restrict__ blockScan, int* __restrict__ rowPtr,
    int* __restrict__ cursor) {
  int gid = blockIdx.x * 256 + threadIdx.x;
  if (gid < NNODES) {
    int b = gid >> 8;
    int off = (b > 0) ? blockScan[b - 1] : 0;
    int ex = incl[gid] + off - deg[gid];   // exclusive prefix
    rowPtr[gid] = ex;
    cursor[gid] = ex;
  }
  if (gid == 0) rowPtr[NNODES] = NEDGES;
}

__global__ __launch_bounds__(256) void gin_csr_fill_kernel(
    const int* __restrict__ src, const int* __restrict__ dst,
    int* __restrict__ cursor, int* __restrict__ colIdx) {
  int e = blockIdx.x * 256 + threadIdx.x;
  if (e >= NEDGES) return;
  int pos = atomicAdd(&cursor[dst[e]], 1);
  colIdx[pos] = src[e];
}

// ---------------------------------------------------------------------------
// Gather: hsum[n] = h[n] + sum_{nbr in CSR[n]} h[nbr]
// One wave per node; lanes own 4 columns -> 512B coalesced reads; no atomics.
// ---------------------------------------------------------------------------
__global__ __launch_bounds__(256) void gin_gather_kernel(
    const float* __restrict__ h, const int* __restrict__ rowPtr,
    const int* __restrict__ colIdx, float* __restrict__ hsum) {
  int n = (blockIdx.x * 256 + threadIdx.x) >> 5;   // node id (wave-uniform)
  if (n >= NNODES) return;
  int lane = threadIdx.x & 31;
  int c = lane << 2;

  v4f acc = *(const v4f*)(h + (size_t)n * D + c);  // (1+eps)*h, eps=0
  int j = rowPtr[n];
  int end = rowPtr[n + 1];
  for (; j + 1 < end; j += 2) {
    int n0 = colIdx[j];
    int n1 = colIdx[j + 1];
    v4f t0 = *(const v4f*)(h + (size_t)n0 * D + c);
    v4f t1 = *(const v4f*)(h + (size_t)n1 * D + c);
    acc += t0;
    acc += t1;
  }
  if (j < end) {
    int n0 = colIdx[j];
    acc += *(const v4f*)(h + (size_t)n0 * D + c);
  }
  *(v4f*)(hsum + (size_t)n * D + c) = acc;
}

// ---------------------------------------------------------------------------
// Fused GIN MLP layer:
//   hout = ReLU( ReLU(hsum @ W1 + b1) @ W2 + b2 ), pool[batch] += hout
// 256 threads = 8 wave32; each wave computes a 16-row x 128-col tile with
// v_wmma_f32_16x16x32_bf16 (4 K-iters x 8 N-tiles per GEMM).
// ---------------------------------------------------------------------------
__global__ __launch_bounds__(256) void gin_mlp_kernel(
    const float* __restrict__ hsum,
    const __bf16* __restrict__ w1t, const __bf16* __restrict__ w2t,
    const float* __restrict__ b1, const float* __restrict__ b2,
    const int* __restrict__ batch,
    float* __restrict__ hout, float* __restrict__ pool, int poolOff) {
  __shared__ __align__(16) __bf16 sW1[D * D];            // 32 KB
  __shared__ __align__(16) __bf16 sW2[D * D];            // 32 KB
  __shared__ __align__(16) __bf16 sStage[8][16 * D];     // 32 KB

  const int tid = threadIdx.x;

  // Cooperative LDS load of both weight matrices.
  {
    const unsigned long long* a = (const unsigned long long*)w1t;
    const unsigned long long* b = (const unsigned long long*)w2t;
    unsigned long long* sa = (unsigned long long*)sW1;
    unsigned long long* sb = (unsigned long long*)sW2;
#pragma unroll
    for (int i = 0; i < (D * D / 4) / 256; ++i) {
      sa[tid + i * 256] = a[tid + i * 256];
      sb[tid + i * 256] = b[tid + i * 256];
    }
  }
  __syncthreads();

  const int wave = tid >> 5;
  const int lane = tid & 31;
  const int sub  = lane >> 4;
  const int lrow = lane & 15;
  const int m0   = blockIdx.x * 128 + wave * 16;

  // ---------------- GEMM1 A-fragments from hsum ----------------------------
  v16bf A1[4];
  {
    int row = m0 + lrow;
    if (row > NNODES - 1) row = NNODES - 1;   // clamp; masked on store
    const float* hp = hsum + (size_t)row * D;
#pragma unroll
    for (int t = 0; t < 4; ++t) {
      const int kA = t * 32 + 16 * sub;
      v4f h0 = *(const v4f*)(hp + kA);
      v4f h1 = *(const v4f*)(hp + kA + 4);
      v4f h2 = *(const v4f*)(hp + kA + 8);
      v4f h3 = *(const v4f*)(hp + kA + 12);
#pragma unroll
      for (int j = 0; j < 4; ++j) {
        A1[t][j]      = (__bf16)h0[j];
        A1[t][4 + j]  = (__bf16)h1[j];
        A1[t][8 + j]  = (__bf16)h2[j];
        A1[t][12 + j] = (__bf16)h3[j];
      }
    }
  }

  v8f acc[8];
#pragma unroll
  for (int nt = 0; nt < 8; ++nt)
#pragma unroll
    for (int v = 0; v < 8; ++v) acc[nt][v] = 0.f;

#pragma unroll
  for (int t = 0; t < 4; ++t) {
#pragma unroll
    for (int nt = 0; nt < 8; ++nt) {
      const __bf16* bp = &sW1[(nt * 16 + lrow) * D + t * 32 + 16 * sub];
      v16bf B = *(const v16bf*)bp;
      acc[nt] = __builtin_amdgcn_wmma_f32_16x16x32_bf16(
          false, A1[t], false, B, (short)0, acc[nt], false, false);
    }
  }

  // bias + ReLU -> bf16 stage tile (cross-lane transpose via LDS)
  __bf16* st = &sStage[wave][0];
#pragma unroll
  for (int nt = 0; nt < 8; ++nt) {
    float bb = b1[nt * 16 + lrow];
#pragma unroll
    for (int v = 0; v < 8; ++v) {
      float x = acc[nt][v] + bb;
      x = x > 0.f ? x : 0.f;
      st[(v + 8 * sub) * D + nt * 16 + lrow] = (__bf16)x;
    }
  }

  // ---------------- GEMM2 --------------------------------------------------
  v8f acc2[8];
#pragma unroll
  for (int nt = 0; nt < 8; ++nt)
#pragma unroll
    for (int v = 0; v < 8; ++v) acc2[nt][v] = 0.f;

#pragma unroll
  for (int t = 0; t < 4; ++t) {
    v16bf A2;
    {
      const __bf16* sp = st + lrow * D;
      const int kA = t * 32 + 16 * sub;
      v8bf lo = *(const v8bf*)(sp + kA);
      v8bf hi = *(const v8bf*)(sp + kA + 8);
#pragma unroll
      for (int j = 0; j < 8; ++j) {
        A2[j] = lo[j];
        A2[8 + j] = hi[j];
      }
    }
#pragma unroll
    for (int nt = 0; nt < 8; ++nt) {
      const __bf16* bp = &sW2[(nt * 16 + lrow) * D + t * 32 + 16 * sub];
      v16bf B = *(const v16bf*)bp;
      acc2[nt] = __builtin_amdgcn_wmma_f32_16x16x32_bf16(
          false, A2, false, B, (short)0, acc2[nt], false, false);
    }
  }

  // ------------- bias + outer ReLU + store h + pooled atomics --------------
  // Lane owns 8 consecutive rows r0..r0+7; batch is sorted, so if endpoints
  // match, one reduced atomic per (lane, n-tile) instead of 8.
  const int r0 = m0 + 8 * sub;
  const int r7 = r0 + 7;
  int gb[8];
#pragma unroll
  for (int v = 0; v < 8; ++v) {
    int rc = (r0 + v) < NNODES ? (r0 + v) : (NNODES - 1);
    gb[v] = batch[rc];
  }
  const bool uniform = (r7 < NNODES) && (gb[0] == gb[7]);

#pragma unroll
  for (int nt = 0; nt < 8; ++nt) {
    float bb = b2[nt * 16 + lrow];
    int col = nt * 16 + lrow;
    float xs[8];
#pragma unroll
    for (int v = 0; v < 8; ++v) {
      float x = acc2[nt][v] + bb;
      xs[v] = x > 0.f ? x : 0.f;
      if (r0 + v < NNODES) hout[(size_t)(r0 + v) * D + col] = xs[v];
    }
    if (uniform) {
      float s = ((xs[0] + xs[1]) + (xs[2] + xs[3])) +
                ((xs[4] + xs[5]) + (xs[6] + xs[7]));
      atomic_add_f32_dev(&pool[gb[0] * (NLAYERS * D) + poolOff + col], s);
    } else {
#pragma unroll
      for (int v = 0; v < 8; ++v) {
        if (r0 + v < NNODES)
          atomic_add_f32_dev(&pool[gb[v] * (NLAYERS * D) + poolOff + col], xs[v]);
      }
    }
  }
}

// ---------------------------------------------------------------------------
// Launch
// ---------------------------------------------------------------------------
static inline char* ws_align(char* p) {
  return (char*)(((uintptr_t)p + 255) & ~(uintptr_t)255);
}

extern "C" void kernel_launch(void* const* d_in, const int* in_sizes, int n_in,
                              void* d_out, int out_size, void* d_ws, size_t ws_size,
                              hipStream_t stream) {
  (void)in_sizes; (void)n_in; (void)out_size; (void)ws_size;

  const float* x     = (const float*)d_in[0];
  const int*   ei    = (const int*)d_in[1];
  const int*   batch = (const int*)d_in[2];
  const float* W1    = (const float*)d_in[3];
  const float* b1    = (const float*)d_in[4];
  const float* W2    = (const float*)d_in[5];
  const float* b2    = (const float*)d_in[6];
  float* pool = (float*)d_out;

  const int* src = ei;            // edge_index[0]
  const int* dst = ei + NEDGES;   // edge_index[1]

  const int NBLK = (NNODES + 255) / 256;   // 196 scan blocks

  // Workspace partition
  char* p = (char*)d_ws;
  __bf16* wt1   = (__bf16*)p; p = ws_align(p + (size_t)NLAYERS * D * D * sizeof(__bf16));
  __bf16* wt2   = (__bf16*)p; p = ws_align(p + (size_t)NLAYERS * D * D * sizeof(__bf16));
  int* deg      = (int*)p;    p = ws_align(p + (size_t)NNODES * sizeof(int));
  int* incl     = (int*)p;    p = ws_align(p + (size_t)NNODES * sizeof(int));
  int* rowPtr   = (int*)p;    p = ws_align(p + (size_t)(NNODES + 1) * sizeof(int));
  int* cursor   = (int*)p;    p = ws_align(p + (size_t)NNODES * sizeof(int));
  int* bsums    = (int*)p;    p = ws_align(p + 256 * sizeof(int));
  int* colIdx   = (int*)p;    p = ws_align(p + (size_t)NEDGES * sizeof(int));
  float* hsum   = (float*)p;  p = ws_align(p + (size_t)NNODES * D * sizeof(float));
  float* hA     = (float*)p;  p = ws_align(p + (size_t)NNODES * D * sizeof(float));
  float* hB     = (float*)p;

  // --- one-time per call: weight pack + CSR build --------------------------
  gin_pack_w_kernel<<<(NLAYERS * D * D + 255) / 256, 256, 0, stream>>>(W1, W2, wt1, wt2);

  gin_zero4_kernel<<<(NNODES / 4 + 255) / 256, 256, 0, stream>>>((v4f*)deg, NNODES / 4);
  gin_count_kernel<<<(NEDGES + 255) / 256, 256, 0, stream>>>(dst, deg);
  gin_scan_block_kernel<<<NBLK, 256, 0, stream>>>(deg, incl, bsums);
  gin_scan_sums_kernel<<<1, 256, 0, stream>>>(bsums, NBLK);
  gin_finalize_rowptr_kernel<<<NBLK, 256, 0, stream>>>(deg, incl, bsums, rowPtr, cursor);
  gin_csr_fill_kernel<<<(NEDGES + 255) / 256, 256, 0, stream>>>(src, dst, cursor, colIdx);

  // Zero pooled output (d_out is poisoned by the harness)
  {
    int n4 = NGRAPHS * NLAYERS * D / 4;
    gin_zero4_kernel<<<(n4 + 255) / 256, 256, 0, stream>>>((v4f*)pool, n4);
  }

  // --- layers ---------------------------------------------------------------
  const float* hin = x;
  float* bufs[2] = {hA, hB};
  for (int l = 0; l < NLAYERS; ++l) {
    gin_gather_kernel<<<(NNODES * 32 + 255) / 256, 256, 0, stream>>>(hin, rowPtr, colIdx, hsum);

    float* ho = bufs[l & 1];
    gin_mlp_kernel<<<(NNODES + 127) / 128, 256, 0, stream>>>(
        hsum, wt1 + (size_t)l * D * D, wt2 + (size_t)l * D * D,
        b1 + l * D, b2 + l * D, batch, ho, pool, l * D);
    hin = ho;
  }
}
